// MFN_20976620274296
// MI455X (gfx1250) — matrix operand order repaired and checked by reading, not compiled, not run on hardware
//
#include <hip/hip_runtime.h>

// ---------------------------------------------------------------------------
// MFN recurrent network on MI455X (gfx1250), wave32 + WMMA bf16.
// Each wave owns a 16-row batch stripe and runs the full T=128 recurrence
// privately (rows are independent). All GEMM operands are kept pre-swizzled
// in WMMA fragment order: A operands in LDS (bf16, ISA 7.12.2 16-bit 16x32
// layout, 1KB/tile), B operands in workspace (bf16, B-fragment layout,
// prepped once per call). Every fragment load is one contiguous 32B load.
// The k-loop is 2-stage software pipelined and processes 2 n-tiles per pass
// (one A fragment feeds two WMMAs).
// ---------------------------------------------------------------------------

#define T_STEPS   128
#define N_BATCH   2048
#define D_LINGU   300
#define D_ACOUS   74
#define D_IMAGE   35
#define H_LINGU   88
#define H_ACOUS   48
#define H_IMAGE   88
#define MEMD      128
#define ATT_INW   448
#define GATEPITCH 352

typedef __attribute__((ext_vector_type(16))) __bf16 v16bf;
typedef __attribute__((ext_vector_type(8)))  float  v8f;

#define ACT_NONE 0
#define ACT_RELU 1
#define ACT_SIGM 2
#define ACT_TANH 3

// ------------------------- LDS layout (bytes) ------------------------------
#define SW_HXL    0u            // [x_lin pad320 | h_l pad96]  : 13 tiles
#define SW_HXA    13312u        // [x_ac pad96  | h_a pad64]   : 5 tiles
#define SW_HXI    18432u        // [x_im pad64  | h_i pad96]   : 5 tiles
#define SW_CSTAR  23552u        // c_star 448                  : 14 tiles
#define SW_MEM    37888u        // mem 128 (MUST follow CSTAR) : 4 tiles
#define SW_HIDS   41984u        // MLP hiddens (<=256)         : 8 tiles
#define LIN_GATES 50176u        // 16 x 352 f32
#define LIN_Z     72704u        // 16 x 448 f32
#define LIN_CL    101376u       // 16 x 88 f32
#define LIN_CA    107008u       // 16 x 48 f32
#define LIN_CI    110080u       // 16 x 88 f32
#define LIN_INV   115712u       // 16 f32 (softmax 1/sum)
#define LDS_BYTES 115776u

// ------------------ workspace (prepped bf16 B matrices) --------------------
#define WOFF_LIN   0
#define WOFF_AC    146432
#define WOFF_IM    177152
#define WOFF_A1A   233472
#define WOFF_A1B   290816
#define WOFF_A2A   348160
#define WOFF_A2B   462848
#define WOFF_G1A   495616
#define WOFF_G1B   532480
#define WOFF_G2A   540672
#define WOFF_G2B   577536
#define WOFF_OUT1  585728

// ---------------------------------------------------------------------------
__device__ __forceinline__ unsigned short f2bfu(float f) {
  union { float fv; unsigned u; } x; x.fv = f;
  unsigned r = x.u + 0x7FFFu + ((x.u >> 16) & 1u);   // round-to-nearest-even
  return (unsigned short)(r >> 16);
}
__device__ __forceinline__ __bf16 f2bf(float f) {
  union { unsigned short s; __bf16 b; } y; y.s = f2bfu(f); return y.b;
}
__device__ __forceinline__ float bfu2f(unsigned bits16) {
  union { unsigned u; float f; } y; y.u = bits16 << 16;
  return y.f;
}
__device__ __forceinline__ float bf2f(__bf16 b) {
  union { unsigned short s; __bf16 b; } x; x.b = b;
  return bfu2f((unsigned)x.s);
}
template <int ACT>
__device__ __forceinline__ float applyAct(float v) {
  if (ACT == ACT_RELU) return fmaxf(v, 0.f);
  if (ACT == ACT_SIGM) return 1.f / (1.f + __expf(-v));
  if (ACT == ACT_TANH) return tanhf(v);
  return v;
}
// element index (bf16 units) of (row m, col k) inside a swizzled A buffer
__device__ __forceinline__ unsigned swzA(int m, int k) {
  int tile = k >> 5, ko = k & 31;
  int lanep = (m & 15) | (((ko >> 3) & 1) << 4);
  int e = (ko & 7) | ((ko >> 4) << 3);
  return (unsigned)(tile * 512 + lanep * 16 + e);
}
__device__ __forceinline__ v16bf ld_v16bf(const void* p) {
  struct U { uint4 a, b; };
  union { U u; v16bf v; } c;
  c.u = *(const U*)p;   // two 16B loads (ds_load_b128 / global_load_b128)
  return c.v;
}
__device__ __forceinline__ v8f wmma_bf16(v16bf a, v16bf b, v8f c) {
  return __builtin_amdgcn_wmma_f32_16x16x32_bf16(false, a, false, b,
                                                 (short)0, c, false, false);
}

// ------------------------------ prep kernel --------------------------------
struct PrepSeg { const float* src; int ld; int colOff; int width; int ptiles; };
struct PrepJob { PrepSeg s[4]; int nseg; int Ntiles; int KT; };

__global__ void prep_bswz(PrepJob j, __bf16* dst) {
  int idx = blockIdx.x * 256 + threadIdx.x;
  int total = j.Ntiles * j.KT * 512;
  if (idx >= total) return;
  int block = idx >> 9;                  // ntile*KT + kt
  int r = idx & 511;
  int lane = r >> 4, e = r & 15;
  int ntile = block / j.KT, kt = block - ntile * j.KT;
  int n = ntile * 16 + (lane & 15);
  int kp = kt * 32 + ((lane >> 4) << 3) + (e & 7) + ((e >> 3) << 4);   // padded-K index
  float v = 0.f;
#pragma unroll
  for (int s = 0; s < 4; ++s) {
    if (s < j.nseg) {
      int pw = j.s[s].ptiles << 5;
      if (kp >= 0 && kp < pw && kp < j.s[s].width)
        v = j.s[s].src[(size_t)n * j.s[s].ld + j.s[s].colOff + kp];
      kp -= pw;
    }
  }
  dst[idx] = f2bf(v);
}

// ------------------------------ main kernel --------------------------------
struct Params {
  const float *x_lin, *x_ac, *x_im;
  const float *b_ih_lin, *b_hh_lin, *b_ih_ac, *b_hh_ac, *b_ih_im, *b_hh_im;
  const float *att1_b1, *att1_b2, *att2_b1, *att2_b2;
  const float *g1_b1, *g1_b2, *g2_b1, *g2_b2;
  const float *out_b1, *out_b2, *out_w2;
  const __bf16* bw;
  float* out;
};

struct ATiles { unsigned off; int kt; };

// EPI: 0 = linear f32 store, 1 = swizzled bf16 store (k = n)
template <int EPI, int ACT>
__device__ __forceinline__ void epilogue(char* lds, float* outLin, int outPitch,
                                         int outColOff, unsigned outSwzOff,
                                         int n, int half, v8f acc, float bv) {
  if (EPI == 0) {
#pragma unroll
    for (int j = 0; j < 8; ++j) {
      int m = j + (half << 3);
      outLin[m * outPitch + outColOff + n] = applyAct<ACT>(acc[j] + bv);
    }
  } else {
    __bf16* dst = (__bf16*)(lds + outSwzOff);
    int tile = n >> 5;
    int hb = ((n >> 3) & 1) << 4;
    int e = (n & 7) | (((n >> 4) & 1) << 3);
    unsigned base = (unsigned)tile * 512u + (unsigned)e;
#pragma unroll
    for (int j = 0; j < 8; ++j) {
      int m = j + (half << 3);
      dst[base + (unsigned)((m & 15) | hb) * 16u] = f2bf(applyAct<ACT>(acc[j] + bv));
    }
  }
}

template <int NA, int EPI, int ACT, bool HASB1>
__device__ __forceinline__ void gemm(char* lds, const ATiles (&as)[NA], int KT,
                                     const __bf16* B, int Ntiles,
                                     const float* bias0, const float* bias1,
                                     float* outLin, int outPitch, int outColOff,
                                     unsigned outSwzOff, int lane) {
  const int half = lane >> 4, lmod = lane & 15;
#pragma unroll 1
  for (int nt = 0; nt < Ntiles; nt += 2) {     // all Ntiles are even
    v8f acc0 = {0.f, 0.f, 0.f, 0.f, 0.f, 0.f, 0.f, 0.f};
    v8f acc1 = {0.f, 0.f, 0.f, 0.f, 0.f, 0.f, 0.f, 0.f};
    int kg = 0;
#pragma unroll
    for (int s = 0; s < NA; ++s) {
      const char* ap = lds + as[s].off + (unsigned)lane * 32u;
      const char* bp = (const char*)B +
          (((size_t)(nt * KT + kg)) * 512u + (unsigned)lane * 16u) * 2u;
      const unsigned bstride = (unsigned)KT * 1024u;  // bytes between n-tiles
      const int kts = as[s].kt;
      // 2-stage pipeline: fragments for kt are loaded one iteration ahead
      v16bf a  = ld_v16bf(ap);
      v16bf b0 = ld_v16bf(bp);
      v16bf b1 = ld_v16bf(bp + bstride);
#pragma unroll 1
      for (int kt = 1; kt < kts; ++kt) {
        v16bf an  = ld_v16bf(ap + (size_t)kt * 1024u);
        v16bf b0n = ld_v16bf(bp + (size_t)kt * 1024u);
        v16bf b1n = ld_v16bf(bp + (size_t)kt * 1024u + bstride);
        acc0 = wmma_bf16(a, b0, acc0);
        acc1 = wmma_bf16(a, b1, acc1);
        a = an; b0 = b0n; b1 = b1n;
      }
      acc0 = wmma_bf16(a, b0, acc0);
      acc1 = wmma_bf16(a, b1, acc1);
      kg += kts;
    }
    const int n0 = nt * 16 + lmod;
    float bv0 = bias0[n0]      + (HASB1 ? bias1[n0]      : 0.f);
    float bv1 = bias0[n0 + 16] + (HASB1 ? bias1[n0 + 16] : 0.f);
    epilogue<EPI, ACT>(lds, outLin, outPitch, outColOff, outSwzOff, n0,      half, acc0, bv0);
    epilogue<EPI, ACT>(lds, outLin, outPitch, outColOff, outSwzOff, n0 + 16, half, acc1, bv1);
  }
}

// stage a 16 x WIDTH f32 global tile into swizzled bf16 LDS, zero-padded to KP
template <int WIDTH, int KP>
__device__ __forceinline__ void stage_x(char* lds, unsigned dstOff, const float* src,
                                        int lane) {
  __bf16* dst = (__bf16*)(lds + dstOff);
  const int rpr = KP >> 3;               // 8-element runs per row (constant)
  const int runs = rpr << 4;
#pragma unroll 1
  for (int ridx = lane; ridx < runs; ridx += 32) {
    int m = ridx / rpr, rk = ridx - m * rpr;
    int k0 = rk << 3;
    const float* sp = src + (size_t)m * WIDTH;
    unsigned o = swzA(m, k0);            // 16B-aligned run of 8 bf16
    unsigned q[4];
#pragma unroll
    for (int i = 0; i < 4; ++i) {
      int ka = k0 + 2 * i, kb = ka + 1;
      float va = sp[ka < WIDTH ? ka : WIDTH - 1];
      float vb = sp[kb < WIDTH ? kb : WIDTH - 1];
      unsigned ha  = (ka < WIDTH) ? (unsigned)f2bfu(va) : 0u;
      unsigned hbv = (kb < WIDTH) ? (unsigned)f2bfu(vb) : 0u;
      q[i] = ha | (hbv << 16);
    }
    *(uint4*)(dst + o) = make_uint4(q[0], q[1], q[2], q[3]);
  }
}

// LSTM elementwise: gates linear f32; c linear f32; h -> swizzled bf16;
// new c -> CSTAR swizzled bf16 at column offset csOff.
template <int H>
__device__ __forceinline__ void lstm_update(char* lds, unsigned cOff,
                                            unsigned hBufOff, int hK0, int csOff,
                                            int lane) {
  const float* gates = (const float*)(lds + LIN_GATES);
  float* c = (float*)(lds + cOff);
  __bf16* hx = (__bf16*)(lds + hBufOff);
  __bf16* cb = (__bf16*)(lds + SW_CSTAR);
#pragma unroll 1
  for (int idx = lane; idx < 16 * H; idx += 32) {
    int r = idx / H, u = idx - r * H;
    const float* g = gates + r * GATEPITCH;
    float ig = 1.f / (1.f + __expf(-g[u]));
    float fg = 1.f / (1.f + __expf(-g[H + u]));
    float gg = tanhf(g[2 * H + u]);
    float og = 1.f / (1.f + __expf(-g[3 * H + u]));
    float c2 = fg * c[r * H + u] + ig * gg;
    c[r * H + u] = c2;
    hx[swzA(r, hK0 + u)] = f2bf(og * tanhf(c2));
    cb[swzA(r, csOff + u)] = f2bf(c2);
  }
}

template <int H>
__device__ __forceinline__ void copy_prev_c(char* lds, unsigned cOff, int csOff, int lane) {
  const float* c = (const float*)(lds + cOff);
  __bf16* cb = (__bf16*)(lds + SW_CSTAR);
#pragma unroll 1
  for (int idx = lane; idx < 16 * H; idx += 32) {
    int r = idx / H, u = idx - r * H;
    cb[swzA(r, csOff + u)] = f2bf(c[r * H + u]);
  }
}

__global__ void __launch_bounds__(32) mfn_kernel(Params p) {
  extern __shared__ char lds[];
  const int lane = threadIdx.x & 31;
  const int row0 = blockIdx.x * 16;

  // zero everything once (states; swizzle pad columns stay zero forever)
  for (unsigned i = lane; i < (LDS_BYTES >> 2); i += 32) ((unsigned*)lds)[i] = 0u;

  float* gatesLin = (float*)(lds + LIN_GATES);
  float* zLin     = (float*)(lds + LIN_Z);
  float* rowInv   = (float*)(lds + LIN_INV);

#pragma unroll 1
  for (int t = 0; t < T_STEPS; ++t) {
    // ---- prefetch next step's x tiles (global_prefetch_b8) ----------------
    if (t + 1 < T_STEPS) {
      const float* nl = p.x_lin + (size_t)((t + 1) * N_BATCH + row0) * D_LINGU;
      const float* na = p.x_ac  + (size_t)((t + 1) * N_BATCH + row0) * D_ACOUS;
      const float* ni = p.x_im  + (size_t)((t + 1) * N_BATCH + row0) * D_IMAGE;
      for (int o = lane * 64; o < 16 * D_LINGU; o += 32 * 64) __builtin_prefetch(nl + o, 0, 0);
      for (int o = lane * 64; o < 16 * D_ACOUS; o += 32 * 64) __builtin_prefetch(na + o, 0, 0);
      for (int o = lane * 64; o < 16 * D_IMAGE; o += 32 * 64) __builtin_prefetch(ni + o, 0, 0);
    }

    // ---- stage x tiles into swizzled bf16 ---------------------------------
    stage_x<D_LINGU, 320>(lds, SW_HXL, p.x_lin + (size_t)(t * N_BATCH + row0) * D_LINGU, lane);
    stage_x<D_ACOUS,  96>(lds, SW_HXA, p.x_ac  + (size_t)(t * N_BATCH + row0) * D_ACOUS, lane);
    stage_x<D_IMAGE,  64>(lds, SW_HXI, p.x_im  + (size_t)(t * N_BATCH + row0) * D_IMAGE, lane);

    // ---- prev cell states -> c_star[:, 0:224] -----------------------------
    copy_prev_c<H_LINGU>(lds, LIN_CL, 0,   lane);
    copy_prev_c<H_ACOUS>(lds, LIN_CA, 88,  lane);
    copy_prev_c<H_IMAGE>(lds, LIN_CI, 136, lane);

    // ---- three LSTMs: gates = [x|h] @ [W_ih|W_hh]^T + b -------------------
    { ATiles a[1] = {{SW_HXL, 13}};
      gemm<1, 0, ACT_NONE, true>(lds, a, 13, p.bw + WOFF_LIN, 22,
           p.b_ih_lin, p.b_hh_lin, gatesLin, GATEPITCH, 0, 0u, lane); }
    lstm_update<H_LINGU>(lds, LIN_CL, SW_HXL, 320, 224, lane);

    { ATiles a[1] = {{SW_HXA, 5}};
      gemm<1, 0, ACT_NONE, true>(lds, a, 5, p.bw + WOFF_AC, 12,
           p.b_ih_ac, p.b_hh_ac, gatesLin, GATEPITCH, 0, 0u, lane); }
    lstm_update<H_ACOUS>(lds, LIN_CA, SW_HXA, 96, 312, lane);

    { ATiles a[1] = {{SW_HXI, 5}};
      gemm<1, 0, ACT_NONE, true>(lds, a, 5, p.bw + WOFF_IM, 22,
           p.b_ih_im, p.b_hh_im, gatesLin, GATEPITCH, 0, 0u, lane); }
    lstm_update<H_IMAGE>(lds, LIN_CI, SW_HXI, 64, 360, lane);

    // ---- attention scores: z = att1_w2 @ relu(att1_w1 @ c_star) -----------
    { ATiles a[1] = {{SW_CSTAR, 14}};
      gemm<1, 1, ACT_RELU, false>(lds, a, 14, p.bw + WOFF_A1A, 8,
           p.att1_b1, nullptr, nullptr, 0, 0, SW_HIDS, lane); }
    { ATiles a[1] = {{SW_HIDS, 4}};
      gemm<1, 0, ACT_NONE, false>(lds, a, 4, p.bw + WOFF_A1B, 28,
           p.att1_b2, nullptr, zLin, ATT_INW, 0, 0u, lane); }

    // ---- softmax over 448; attended = attn * c_star (swizzled, in place) --
    {
      const int r = lane >> 1, hh = lane & 1;
      float* zr = zLin + r * ATT_INW;
      const int k0 = hh * 224, k1 = k0 + 224;
      float mx = -3.0e38f;
      for (int k = k0; k < k1; ++k) mx = fmaxf(mx, zr[k]);
      mx = fmaxf(mx, __shfl_xor(mx, 1, 32));
      float sum = 0.f;
      for (int k = k0; k < k1; ++k) { float e = __expf(zr[k] - mx); zr[k] = e; sum += e; }
      sum += __shfl_xor(sum, 1, 32);
      if (hh == 0) rowInv[r] = 1.f / sum;
    }
    {
      unsigned* cw = (unsigned*)(lds + SW_CSTAR);     // 2 bf16 per u32
#pragma unroll 1
      for (int w = lane; w < 14 * 256; w += 32) {
        int tile = w >> 8, r = w & 255, lp = r >> 3, ew = r & 7;
        int m = lp & 15, e = ew << 1;
        int k = (tile << 5) + ((lp >> 4) << 3) + (e & 7) + ((e >> 3) << 4);
        unsigned wv = cw[w];
        float inv = rowInv[m];
        float a0 = zLin[m * ATT_INW + k] * inv;
        float a1 = zLin[m * ATT_INW + k + 1] * inv;
        unsigned r0 = f2bfu(bfu2f(wv & 0xffffu) * a0);
        unsigned r1 = f2bfu(bfu2f(wv >> 16) * a1);
        cw[w] = r0 | (r1 << 16);
      }
    }

    // ---- c_hat = tanh(att2(attended)) -> zLin[:, 0:128] -------------------
    { ATiles a[1] = {{SW_CSTAR, 14}};
      gemm<1, 1, ACT_RELU, false>(lds, a, 14, p.bw + WOFF_A2A, 16,
           p.att2_b1, nullptr, nullptr, 0, 0, SW_HIDS, lane); }
    { ATiles a[1] = {{SW_HIDS, 8}};
      gemm<1, 0, ACT_TANH, false>(lds, a, 8, p.bw + WOFF_A2B, 8,
           p.att2_b2, nullptr, zLin, ATT_INW, 0, 0u, lane); }

    // ---- gamma1 = sigm(g1([attended|mem])) -> zLin[:, 128:256] ------------
    { ATiles a[1] = {{SW_CSTAR, 18}};                   // CSTAR+MEM contiguous
      gemm<1, 1, ACT_RELU, false>(lds, a, 18, p.bw + WOFF_G1A, 4,
           p.g1_b1, nullptr, nullptr, 0, 0, SW_HIDS, lane); }
    { ATiles a[1] = {{SW_HIDS, 2}};
      gemm<1, 0, ACT_SIGM, false>(lds, a, 2, p.bw + WOFF_G1B, 8,
           p.g1_b2, nullptr, zLin, ATT_INW, 128, 0u, lane); }
    // ---- gamma2 -> zLin[:, 256:384] ---------------------------------------
    { ATiles a[1] = {{SW_CSTAR, 18}};
      gemm<1, 1, ACT_RELU, false>(lds, a, 18, p.bw + WOFF_G2A, 4,
           p.g2_b1, nullptr, nullptr, 0, 0, SW_HIDS, lane); }
    { ATiles a[1] = {{SW_HIDS, 2}};
      gemm<1, 0, ACT_SIGM, false>(lds, a, 2, p.bw + WOFF_G2B, 8,
           p.g2_b2, nullptr, zLin, ATT_INW, 256, 0u, lane); }

    // ---- mem = gamma1*mem + gamma2*c_hat (swizzled, in place) -------------
    {
      unsigned* mw = (unsigned*)(lds + SW_MEM);
#pragma unroll 1
      for (int w = lane; w < 4 * 256; w += 32) {
        int tile = w >> 8, r = w & 255, lp = r >> 3, ew = r & 7;
        int m = lp & 15, e = ew << 1;
        int k = (tile << 5) + ((lp >> 4) << 3) + (e & 7) + ((e >> 3) << 4);
        unsigned wv = mw[w];
        const float* zm = zLin + m * ATT_INW;
        float n0 = zm[128 + k]     * bfu2f(wv & 0xffffu) + zm[256 + k]     * zm[k];
        float n1 = zm[128 + k + 1] * bfu2f(wv >> 16)     + zm[256 + k + 1] * zm[k + 1];
        mw[w] = (unsigned)f2bfu(n0) | ((unsigned)f2bfu(n1) << 16);
      }
    }

    // ---- output MLP on feats = [h_l|h_a|h_i|mem] --------------------------
    { ATiles a[4] = {{SW_HXL + 10 * 1024u, 3}, {SW_HXA + 3 * 1024u, 2},
                     {SW_HXI + 2 * 1024u, 3}, {SW_MEM, 4}};
      gemm<4, 1, ACT_RELU, false>(lds, a, 12, p.bw + WOFF_OUT1, 4,
           p.out_b1, nullptr, nullptr, 0, 0, SW_HIDS, lane); }
    if (lane < 16) {
      const __bf16* hb = (const __bf16*)(lds + SW_HIDS);
      float acc = p.out_b2[0];
#pragma unroll
      for (int k = 0; k < 64; ++k) acc += bf2f(hb[swzA(lane, k)]) * p.out_w2[k];
      p.out[(size_t)(row0 + lane) * T_STEPS + t] = acc;   // out[row][t]
    }
  }
}

// ------------------------------- host side ---------------------------------
static inline PrepJob mkjob1(const float* s, int ld, int w, int pt, int Nt, int KT) {
  PrepJob j{}; j.s[0] = {s, ld, 0, w, pt}; j.nseg = 1; j.Ntiles = Nt; j.KT = KT; return j;
}

extern "C" void kernel_launch(void* const* d_in, const int* in_sizes, int n_in,
                              void* d_out, int out_size, void* d_ws, size_t ws_size,
                              hipStream_t stream) {
  (void)in_sizes; (void)n_in; (void)out_size; (void)ws_size;
  auto F = [&](int i) { return (const float*)d_in[i]; };
  __bf16* bw = (__bf16*)d_ws;

  PrepJob jobs[12];
  int     joff[12];
  jobs[0] = PrepJob{{{F(3), D_LINGU, 0, D_LINGU, 10}, {F(4), H_LINGU, 0, H_LINGU, 3}},
                    2, 22, 13};                         joff[0] = WOFF_LIN;
  jobs[1] = PrepJob{{{F(7), D_ACOUS, 0, D_ACOUS, 3}, {F(8), H_ACOUS, 0, H_ACOUS, 2}},
                    2, 12, 5};                          joff[1] = WOFF_AC;
  jobs[2] = PrepJob{{{F(11), D_IMAGE, 0, D_IMAGE, 2}, {F(12), H_IMAGE, 0, H_IMAGE, 3}},
                    2, 22, 5};                          joff[2] = WOFF_IM;
  jobs[3] = mkjob1(F(15), 448, 448, 14,  8, 14);        joff[3] = WOFF_A1A;  // att1_w1
  jobs[4] = mkjob1(F(17), 128, 128,  4, 28,  4);        joff[4] = WOFF_A1B;  // att1_w2
  jobs[5] = mkjob1(F(19), 448, 448, 14, 16, 14);        joff[5] = WOFF_A2A;  // att2_w1
  jobs[6] = mkjob1(F(21), 256, 256,  8,  8,  8);        joff[6] = WOFF_A2B;  // att2_w2
  jobs[7] = mkjob1(F(23), 576, 576, 18,  4, 18);        joff[7] = WOFF_G1A;  // g1_w1
  jobs[8] = mkjob1(F(25),  64,  64,  2,  8,  2);        joff[8] = WOFF_G1B;  // g1_w2
  jobs[9] = mkjob1(F(27), 576, 576, 18,  4, 18);        joff[9] = WOFF_G2A;  // g2_w1
  jobs[10] = mkjob1(F(29), 64,  64,  2,  8,  2);        joff[10] = WOFF_G2B; // g2_w2
  jobs[11] = PrepJob{{{F(31), 352, 0, 88, 3}, {F(31), 352, 88, 48, 2},
                      {F(31), 352, 136, 88, 3}, {F(31), 352, 224, 128, 4}},
                     4, 4, 12};                         joff[11] = WOFF_OUT1;

  for (int i = 0; i < 12; ++i) {
    int total = jobs[i].Ntiles * jobs[i].KT * 512;
    hipLaunchKernelGGL(prep_bswz, dim3((total + 255) / 256), dim3(256), 0, stream,
                       jobs[i], bw + joff[i]);
  }

  Params p;
  p.x_lin = F(0); p.x_ac = F(1); p.x_im = F(2);
  p.b_ih_lin = F(5);  p.b_hh_lin = F(6);
  p.b_ih_ac  = F(9);  p.b_hh_ac  = F(10);
  p.b_ih_im  = F(13); p.b_hh_im  = F(14);
  p.att1_b1 = F(16); p.att1_b2 = F(18);
  p.att2_b1 = F(20); p.att2_b2 = F(22);
  p.g1_b1 = F(24); p.g1_b2 = F(26);
  p.g2_b1 = F(28); p.g2_b2 = F(30);
  p.out_b1 = F(32); p.out_w2 = F(33); p.out_b2 = F(34);
  p.bw = bw;
  p.out = (float*)d_out;

  hipLaunchKernelGGL(mfn_kernel, dim3(N_BATCH / 16), dim3(32), LDS_BYTES, stream, p);
}
